// attention_37572373905629
// MI455X (gfx1250) — matrix operand order repaired and checked by reading, not compiled, not run on hardware
//
#include <hip/hip_runtime.h>

typedef __bf16 bf16_t;
typedef __attribute__((ext_vector_type(16))) __bf16 v16bf;
typedef __attribute__((ext_vector_type(8)))  __bf16 v8bf;
typedef __attribute__((ext_vector_type(8)))  float  v8f;

union FragAB { v16bf v; v8bf h[2]; };

// Problem sizes
#define BB  2
#define SS  2048
#define EE  1024
#define HH  16
#define DHD 64
#define NR  4096          // B*S rows

// Workspace element offsets (bf16 elements)
#define OFF_XB 0ull
#define OFF_WQ 4194304ull                  // NR*EE
#define OFF_WK (OFF_WQ + 1048576ull)
#define OFF_WV (OFF_WK + 1048576ull)
#define OFF_WO (OFF_WV + 1048576ull)
#define OFF_Q  (OFF_WO + 1048576ull)       // [B,H,S,DH]
#define OFF_K  (OFF_Q + 4194304ull)        // [B,H,S,DH]
#define OFF_VT (OFF_K + 4194304ull)        // [B,H,DH,S]  (transposed V)
#define OFF_AT (OFF_VT + 4194304ull)       // [B,S,E] attention result

// ---------------------------------------------------------------------------
// Kernel 0: fp32 -> bf16 casts of x and the four weight matrices
// ---------------------------------------------------------------------------
__global__ void cast_inputs(const float* __restrict__ x,
                            const float* __restrict__ wq,
                            const float* __restrict__ wk,
                            const float* __restrict__ wv,
                            const float* __restrict__ wo,
                            bf16_t* __restrict__ ws) {
  unsigned idx = blockIdx.x * blockDim.x + threadIdx.x;
  const unsigned XN = 4194304u, WN = 1048576u;
  if (idx < XN) { ws[OFF_XB + idx] = (bf16_t)x[idx]; return; }
  unsigned r = idx - XN;
  if (r >= 4u * WN) return;
  unsigned seg = r / WN, o = r % WN;
  const float* src = (seg == 0) ? wq : (seg == 1) ? wk : (seg == 2) ? wv : wo;
  ws[OFF_WQ + (unsigned long long)seg * WN + o] = (bf16_t)src[o];
}

// ---------------------------------------------------------------------------
// Kernel 1: fused QKV projection.  out[n,e] = sum_k x[n,k]*W[e,k] + b[e]
// One wave -> one 16x16 tile of Q, K and V (A fragment shared by all three).
// Results stored in the reference's reshaped [B,H,S2,DH] layout; V transposed.
// ---------------------------------------------------------------------------
__global__ void __launch_bounds__(256) qkv_proj(bf16_t* ws,
                                                const float* __restrict__ bq,
                                                const float* __restrict__ bk,
                                                const float* __restrict__ bv) {
  const int lane = threadIdx.x & 31;
  const int w    = blockIdx.x * 8 + (threadIdx.x >> 5);   // 0..16383
  const int m0   = (w >> 6) << 4;                         // row tile (of NR)
  const int n0   = (w & 63) << 4;                         // col tile (of EE)
  const int lr   = lane & 15, hs = lane >> 4;

  const bf16_t* xb = ws + OFF_XB;
  const bf16_t* arow = xb + (unsigned)(m0 + lr) * EE;
  const bf16_t* brow[3] = { ws + OFF_WQ + (unsigned)(n0 + lr) * EE,
                            ws + OFF_WK + (unsigned)(n0 + lr) * EE,
                            ws + OFF_WV + (unsigned)(n0 + lr) * EE };

  v8f acc[3];
  for (int m = 0; m < 3; ++m)
    for (int v = 0; v < 8; ++v) acc[m][v] = 0.f;

  for (int k = 0; k < EE; k += 32) {
    FragAB a;                                   // A: 16x32, lane=row, halves=K
    a.h[0] = *(const v8bf*)(arow + k + 8 * hs);
    a.h[1] = *(const v8bf*)(arow + k + 8 * hs + 16);
#pragma unroll
    for (int m = 0; m < 3; ++m) {
      FragAB b;                                 // B: 32x16, lane=col, halves=K
      b.h[0] = *(const v8bf*)(brow[m] + k + 8 * hs);
      b.h[1] = *(const v8bf*)(brow[m] + k + 8 * hs + 16);
      acc[m] = __builtin_amdgcn_wmma_f32_16x16x32_bf16(
          false, a.v, false, b.v, (short)0, acc[m], false, false);
    }
  }

  const int j = n0 + lr;                        // output column (C: lane=col)
  const float bias[3] = { bq[j], bk[j], bv[j] };
  bf16_t* qb = ws + OFF_Q;
  bf16_t* kb = ws + OFF_K;
  bf16_t* vt = ws + OFF_VT;
#pragma unroll
  for (int v = 0; v < 8; ++v) {
    const int i   = m0 + v + 8 * hs;            // output row
    const int bi  = i >> 11;                    // / S
    const int ii  = i & 2047;
    const int hh  = ii >> 7;                    // / (S/(E/DH)) = /128
    const int rr  = ii & 127;
    const int s2  = (rr << 4) + (j >> 6);       // reshaped sequence index
    const int dd  = j & 63;
    const unsigned long long hb = (unsigned long long)(bi * HH + hh);
    const unsigned long long qi = (hb * SS + s2) * DHD + dd;
    qb[qi] = (bf16_t)(acc[0][v] + bias[0]);
    kb[qi] = (bf16_t)(acc[1][v] + bias[1]);
    vt[(hb * DHD + dd) * SS + s2] = (bf16_t)(acc[2][v] + bias[2]);
  }
}

// ---------------------------------------------------------------------------
// Kernel 2: flash attention.  One wave per (b,h,16-query tile).
// 32 keys/step: 4 WMMA for scores, online softmax (f32), LDS transpose of P,
// 4 WMMA for the PV update.  V is pre-transposed so B-frags are contiguous.
// ---------------------------------------------------------------------------
__global__ void __launch_bounds__(256) attn_fa(bf16_t* ws) {
  __shared__ __align__(16) bf16_t pstage[8][16 * 32];   // per-wave P staging

  const int lane = threadIdx.x & 31;
  const int wvid = threadIdx.x >> 5;
  const int w    = blockIdx.x * 8 + wvid;     // 0..4095
  const int bh   = w >> 7;                    // b*H + h
  const int m0   = (w & 127) << 4;            // query tile base (s2)
  const int lr   = lane & 15, hs = lane >> 4;

  const bf16_t* Q = ws + OFF_Q  + (unsigned long long)bh * SS * DHD;
  const bf16_t* K = ws + OFF_K  + (unsigned long long)bh * SS * DHD;
  const bf16_t* V = ws + OFF_VT + (unsigned long long)bh * DHD * SS;

  FragAB qf[2];                               // Q 16x64 as two K=32 A-frags
  {
    const bf16_t* qrow = Q + (unsigned)(m0 + lr) * DHD;
#pragma unroll
    for (int c = 0; c < 2; ++c) {
      qf[c].h[0] = *(const v8bf*)(qrow + 32 * c + 8 * hs);
      qf[c].h[1] = *(const v8bf*)(qrow + 32 * c + 8 * hs + 16);
    }
  }

  v8f o[4];
  for (int t = 0; t < 4; ++t)
    for (int v = 0; v < 8; ++v) o[t][v] = 0.f;
  float mst[8], lst[8];
  for (int v = 0; v < 8; ++v) { mst[v] = -1e30f; lst[v] = 0.f; }

  for (int kb0 = 0; kb0 < SS; kb0 += 32) {
    v8f s0, s1;
    for (int v = 0; v < 8; ++v) { s0[v] = 0.f; s1[v] = 0.f; }
#pragma unroll
    for (int nt = 0; nt < 2; ++nt) {          // two 16-key column tiles
      const bf16_t* kr = K + (unsigned)(kb0 + 16 * nt + lr) * DHD;
#pragma unroll
      for (int c = 0; c < 2; ++c) {           // d-chunks 0..31, 32..63
        FragAB bk;
        bk.h[0] = *(const v8bf*)(kr + 32 * c + 8 * hs);
        bk.h[1] = *(const v8bf*)(kr + 32 * c + 8 * hs + 16);
        if (nt == 0)
          s0 = __builtin_amdgcn_wmma_f32_16x16x32_bf16(
              false, qf[c].v, false, bk.v, (short)0, s0, false, false);
        else
          s1 = __builtin_amdgcn_wmma_f32_16x16x32_bf16(
              false, qf[c].v, false, bk.v, (short)0, s1, false, false);
      }
    }

    // ---- online softmax (rows live in VGPR index; cols across 16 lanes) ----
    float alpha[8];
#pragma unroll
    for (int v = 0; v < 8; ++v) {
      s0[v] *= 0.125f;                        // 1/sqrt(64)
      s1[v] *= 0.125f;
      float rv = fmaxf(s0[v], s1[v]);
      rv = fmaxf(rv, __shfl_xor(rv, 1, 32));
      rv = fmaxf(rv, __shfl_xor(rv, 2, 32));
      rv = fmaxf(rv, __shfl_xor(rv, 4, 32));
      rv = fmaxf(rv, __shfl_xor(rv, 8, 32));
      const float mn = fmaxf(mst[v], rv);
      alpha[v] = __expf(mst[v] - mn);
      mst[v] = mn;
      s0[v] = __expf(s0[v] - mn);
      s1[v] = __expf(s1[v] - mn);
      float rs = s0[v] + s1[v];
      rs += __shfl_xor(rs, 1, 32);
      rs += __shfl_xor(rs, 2, 32);
      rs += __shfl_xor(rs, 4, 32);
      rs += __shfl_xor(rs, 8, 32);
      lst[v] = lst[v] * alpha[v] + rs;
    }
    for (int t = 0; t < 4; ++t)
      for (int v = 0; v < 8; ++v) o[t][v] *= alpha[v];

    // ---- transpose P (C layout -> A layout) via wave-private LDS ----
    bf16_t* pb = pstage[wvid];
#pragma unroll
    for (int v = 0; v < 8; ++v) {
      const int m = v + 8 * hs;
      pb[m * 32 + lr]      = (bf16_t)s0[v];
      pb[m * 32 + 16 + lr] = (bf16_t)s1[v];
    }
    FragAB pf;                                // P as 16x32 A-frag
    {
      const bf16_t* pr = pb + lr * 32 + 8 * hs;
      pf.h[0] = *(const v8bf*)(pr);
      pf.h[1] = *(const v8bf*)(pr + 16);
    }

    // ---- O += P x V  (V^T rows are contiguous over the key index) ----
#pragma unroll
    for (int t = 0; t < 4; ++t) {
      const bf16_t* vr = V + (unsigned)(16 * t + lr) * SS + kb0 + 8 * hs;
      FragAB bv;
      bv.h[0] = *(const v8bf*)(vr);
      bv.h[1] = *(const v8bf*)(vr + 16);
      o[t] = __builtin_amdgcn_wmma_f32_16x16x32_bf16(
          false, pf.v, false, bv.v, (short)0, o[t], false, false);
    }
  }

  // epilogue: divide by row sums, store in [B,S2,E] (transpose(0,2,1,3) view)
  const int b = bh >> 4, h = bh & 15;
  bf16_t* at = ws + OFF_AT;
#pragma unroll
  for (int v = 0; v < 8; ++v) {
    const float inv = 1.0f / lst[v];
    const int s2 = m0 + v + 8 * hs;
    const unsigned long long rowb =
        ((unsigned long long)(b * SS + s2)) * EE + h * DHD;
    for (int t = 0; t < 4; ++t)
      at[rowb + 16 * t + lr] = (bf16_t)(o[t][v] * inv);
  }
}

// ---------------------------------------------------------------------------
// Kernel 3: output projection, f32 result straight to d_out.
// ---------------------------------------------------------------------------
__global__ void __launch_bounds__(256) out_proj(const bf16_t* __restrict__ ws,
                                                const float* __restrict__ bo,
                                                float* __restrict__ out) {
  const int lane = threadIdx.x & 31;
  const int w    = blockIdx.x * 8 + (threadIdx.x >> 5);
  const int m0   = (w >> 6) << 4;
  const int n0   = (w & 63) << 4;
  const int lr   = lane & 15, hs = lane >> 4;

  const bf16_t* arow = ws + OFF_AT + (unsigned)(m0 + lr) * EE;
  const bf16_t* brow = ws + OFF_WO + (unsigned)(n0 + lr) * EE;

  v8f acc;
  for (int v = 0; v < 8; ++v) acc[v] = 0.f;

  for (int k = 0; k < EE; k += 32) {
    FragAB a, b;
    a.h[0] = *(const v8bf*)(arow + k + 8 * hs);
    a.h[1] = *(const v8bf*)(arow + k + 8 * hs + 16);
    b.h[0] = *(const v8bf*)(brow + k + 8 * hs);
    b.h[1] = *(const v8bf*)(brow + k + 8 * hs + 16);
    acc = __builtin_amdgcn_wmma_f32_16x16x32_bf16(
        false, a.v, false, b.v, (short)0, acc, false, false);
  }

  const int j = n0 + lr;
  const float bias = bo[j];
#pragma unroll
  for (int v = 0; v < 8; ++v) {
    const int i = m0 + v + 8 * hs;
    out[(unsigned long long)i * EE + j] = acc[v] + bias;
  }
}

// ---------------------------------------------------------------------------
extern "C" void kernel_launch(void* const* d_in, const int* in_sizes, int n_in,
                              void* d_out, int out_size, void* d_ws, size_t ws_size,
                              hipStream_t stream) {
  const float* x  = (const float*)d_in[0];
  const float* Wq = (const float*)d_in[1];
  const float* bq = (const float*)d_in[2];
  const float* Wk = (const float*)d_in[3];
  const float* bk = (const float*)d_in[4];
  const float* Wv = (const float*)d_in[5];
  const float* bv = (const float*)d_in[6];
  const float* Wo = (const float*)d_in[7];
  const float* bo = (const float*)d_in[8];
  bf16_t* ws = (bf16_t*)d_ws;
  float* out = (float*)d_out;

  // 0) cast fp32 inputs to bf16 (8.39M elements)
  cast_inputs<<<32768, 256, 0, stream>>>(x, Wq, Wk, Wv, Wo, ws);
  // 1) fused QKV projection: 16384 16x16 tiles, 8 waves/block
  qkv_proj<<<2048, 256, 0, stream>>>(ws, bq, bk, bv);
  // 2) flash attention: 4096 waves (one per b,h,q-tile)
  attn_fa<<<512, 256, 0, stream>>>(ws);
  // 3) output projection -> f32 d_out
  out_proj<<<2048, 256, 0, stream>>>(ws, bo, out);
}
